// NativeMambaBlock_64123861729635
// MI455X (gfx1250) — compile-verified, matrix operand order checked by reading, not run on hardware
//
#include <hip/hip_runtime.h>
#include <hip/hip_bf16.h>
#include <math.h>

// ---------------------------------------------------------------------------
// Types for gfx1250 WMMA (wave32): A/B = v16h (8 VGPRs), C/D = v8f (8 VGPRs)
// ---------------------------------------------------------------------------
typedef _Float16 f16;
typedef __attribute__((ext_vector_type(16))) _Float16 v16h;
typedef __attribute__((ext_vector_type(8)))  _Float16 v8h;
typedef __attribute__((ext_vector_type(8)))  float    v8f;
typedef __attribute__((ext_vector_type(4)))  int      v4i;

static __device__ __forceinline__ float sigmoidf_fast(float x) {
  return 1.0f / (1.0f + __expf(-x));
}

// ---------------------------------------------------------------------------
// Async global->LDS copy (gfx1250 GLOBAL_LOAD_ASYNC_TO_LDS_B128, ASYNCcnt).
// This toolchain declares the builtin with generic `int4*` params.
// Guarded by __has_builtin; falls back to batched reg-staged copy.
// ---------------------------------------------------------------------------
#if defined(__has_builtin)
#if __has_builtin(__builtin_amdgcn_global_load_async_to_lds_b128)
#define ASYNC_COPY 1
#endif
#endif
#ifndef ASYNC_COPY
#define ASYNC_COPY 0
#endif

#if ASYNC_COPY
static __device__ __forceinline__ void copy16B_async(f16* dst, const f16* src) {
  __builtin_amdgcn_global_load_async_to_lds_b128((v4i*)src, (v4i*)dst,
                                                 /*offset=*/0, /*cpol=*/0);
}
static __device__ __forceinline__ void fence_async() {
#if __has_builtin(__builtin_amdgcn_s_wait_asynccnt)
  __builtin_amdgcn_s_wait_asynccnt(0);
#else
  asm volatile("s_wait_asynccnt 0x0" ::: "memory");
#endif
}
#else
static __device__ __forceinline__ void fence_async() {}
#endif

// ---------------------------------------------------------------------------
// Fragment load from LDS per CDNA5 16-bit A/B layout (ISA 7.12.2):
// lane = half*16 + row16 ; elements e=0..7  -> K = 8*half + e
//                          elements e=8..15 -> K = 16 + 8*half + (e-8)
// => two contiguous 16-byte (8 x f16) chunks per lane.
// ---------------------------------------------------------------------------
#define LDSTRIDE 40  // 32 + 8 f16 pad; row start stays 16B-aligned (80 bytes)

union FragU { v16h v; v8h h[2]; };

static __device__ __forceinline__ v16h load_frag16(const f16* rowp, int hlf) {
  FragU u;
  u.h[0] = *(const v8h*)(rowp + 8 * hlf);
  u.h[1] = *(const v8h*)(rowp + 16 + 8 * hlf);
  return u.v;
}

// ---------------------------------------------------------------------------
// WMMA GEMM:  out[m][n] = epi( sum_k A[m][k] * W[n][k] )
//   A: (M x K) f16 row-major, lda
//   W: (N x K) f16 row-major, ldw   (i.e. B = W^T, columns contiguous in K)
// Double-buffered LDS; tile k+1 DMA'd (async-to-LDS) while tile k runs WMMA.
// EPI: 0 = f16 store, 1 = f32 store, 2 = softplus(x + bias[n]) f32,
//      3 = x + aux[m][n] residual f32
// ---------------------------------------------------------------------------
template <int BM, int BN, int WAVES_M, int WAVES_N, int WM_T, int WN_T, int EPI>
__global__ __launch_bounds__(WAVES_M * WAVES_N * 32)
void gemm_wmma_kernel(const f16* __restrict__ A, int lda,
                      const f16* __restrict__ W, int ldw,
                      void* __restrict__ outp, int M, int N, int K,
                      const float* __restrict__ aux, int ldaux) {
  static_assert(BM == WAVES_M * WM_T * 16, "BM mismatch");
  static_assert(BN == WAVES_N * WN_T * 16, "BN mismatch");
  constexpr int NT = WAVES_M * WAVES_N * 32;
  constexpr int CH_A = (BM * 4 + NT - 1) / NT;  // 16B chunks per thread (A)
  constexpr int CH_B = (BN * 4 + NT - 1) / NT;  // 16B chunks per thread (B)
  constexpr bool A_EXACT = (BM * 4) % NT == 0;
  constexpr bool B_EXACT = (BN * 4) % NT == 0;

  __shared__ __align__(16) f16 As[2][BM * LDSTRIDE];
  __shared__ __align__(16) f16 Bs[2][BN * LDSTRIDE];

  const int tid  = threadIdx.x;
  const int lane = tid & 31;
  const int wid  = tid >> 5;
  const int wm   = wid / WAVES_N;
  const int wn   = wid % WAVES_N;
  const int hlf  = lane >> 4;
  const int l16  = lane & 15;

  const int row0 = blockIdx.y * BM;
  const int col0 = blockIdx.x * BN;
  const f16* __restrict__ Aptr = A + (size_t)row0 * lda;
  const f16* __restrict__ Wptr = W + (size_t)col0 * ldw;

  // Stage one 32-deep K-slab of A (BM x 32) and W (BN x 32) into buffer bufi.
  auto stage = [&](int bufi, int kk) {
#if ASYNC_COPY
#pragma unroll
    for (int i = 0; i < CH_A; ++i) {
      int c = tid + i * NT;
      if (A_EXACT || c < BM * 4) {
        int r = c >> 2, cc = (c & 3) << 3;
        copy16B_async(&As[bufi][r * LDSTRIDE + cc],
                      &Aptr[(size_t)r * lda + kk + cc]);
      }
    }
#pragma unroll
    for (int i = 0; i < CH_B; ++i) {
      int c = tid + i * NT;
      if (B_EXACT || c < BN * 4) {
        int r = c >> 2, cc = (c & 3) << 3;
        copy16B_async(&Bs[bufi][r * LDSTRIDE + cc],
                      &Wptr[(size_t)r * ldw + kk + cc]);
      }
    }
#else
    // Batched fallback: issue all global loads, then all LDS stores, so one
    // wait covers every load in flight (no per-chunk serialization).
    v8h ta[CH_A], tb[CH_B];
#pragma unroll
    for (int i = 0; i < CH_A; ++i) {
      int c = tid + i * NT;
      if (A_EXACT || c < BM * 4) {
        int r = c >> 2, cc = (c & 3) << 3;
        ta[i] = *(const v8h*)&Aptr[(size_t)r * lda + kk + cc];
      }
    }
#pragma unroll
    for (int i = 0; i < CH_B; ++i) {
      int c = tid + i * NT;
      if (B_EXACT || c < BN * 4) {
        int r = c >> 2, cc = (c & 3) << 3;
        tb[i] = *(const v8h*)&Wptr[(size_t)r * ldw + kk + cc];
      }
    }
#pragma unroll
    for (int i = 0; i < CH_A; ++i) {
      int c = tid + i * NT;
      if (A_EXACT || c < BM * 4) {
        int r = c >> 2, cc = (c & 3) << 3;
        *(v8h*)&As[bufi][r * LDSTRIDE + cc] = ta[i];
      }
    }
#pragma unroll
    for (int i = 0; i < CH_B; ++i) {
      int c = tid + i * NT;
      if (B_EXACT || c < BN * 4) {
        int r = c >> 2, cc = (c & 3) << 3;
        *(v8h*)&Bs[bufi][r * LDSTRIDE + cc] = tb[i];
      }
    }
#endif
  };

  v8f acc[WM_T][WN_T];
#pragma unroll
  for (int i = 0; i < WM_T; ++i)
#pragma unroll
    for (int j = 0; j < WN_T; ++j)
      acc[i][j] = (v8f){0.f, 0.f, 0.f, 0.f, 0.f, 0.f, 0.f, 0.f};

  stage(0, 0);
  fence_async();
  __syncthreads();

  int buf = 0;
  for (int k0 = 0; k0 < K; k0 += 32) {
    if (k0 + 32 < K) stage(buf ^ 1, k0 + 32);  // DMA next slab during WMMA

    v16h af[WM_T], bf[WN_T];
#pragma unroll
    for (int mt = 0; mt < WM_T; ++mt)
      af[mt] =
          load_frag16(&As[buf][(wm * WM_T * 16 + mt * 16 + l16) * LDSTRIDE], hlf);
#pragma unroll
    for (int nt = 0; nt < WN_T; ++nt)
      bf[nt] =
          load_frag16(&Bs[buf][(wn * WN_T * 16 + nt * 16 + l16) * LDSTRIDE], hlf);

#pragma unroll
    for (int mt = 0; mt < WM_T; ++mt)
#pragma unroll
      for (int nt = 0; nt < WN_T; ++nt)
        acc[mt][nt] = __builtin_amdgcn_wmma_f32_16x16x32_f16(
            false, af[mt], false, bf[nt], (short)0, acc[mt][nt], false, false);

    fence_async();
    __syncthreads();
    buf ^= 1;
  }

  // D layout: VGPR r -> row = r + 8*half, col = lane&15
#pragma unroll
  for (int mt = 0; mt < WM_T; ++mt)
#pragma unroll
    for (int nt = 0; nt < WN_T; ++nt)
#pragma unroll
      for (int r = 0; r < 8; ++r) {
        int gm = row0 + wm * WM_T * 16 + mt * 16 + r + 8 * hlf;
        int gn = col0 + wn * WN_T * 16 + nt * 16 + l16;
        float v = acc[mt][nt][r];
        size_t oi = (size_t)gm * N + gn;
        if (EPI == 0) {
          ((f16*)outp)[oi] = (f16)v;
        } else if (EPI == 1) {
          ((float*)outp)[oi] = v;
        } else if (EPI == 2) {
          v += aux[gn];
          v = (v > 20.f) ? v : log1pf(__expf(v));
          ((float*)outp)[oi] = v;
        } else {
          v += aux[(size_t)gm * ldaux + gn];
          ((float*)outp)[oi] = v;
        }
      }
}

// ---------------------------------------------------------------------------
// LayerNorm over D=1024 + cast to f16. One block per row, 256 threads.
// ---------------------------------------------------------------------------
__global__ __launch_bounds__(256)
void layernorm_cast_kernel(const float* __restrict__ x,
                           const float* __restrict__ w,
                           const float* __restrict__ b,
                           f16* __restrict__ out) {
  const int row = blockIdx.x;
  const float* xr = x + (size_t)row * 1024;
  const int tid = threadIdx.x;

  float v[4], s = 0.f, s2 = 0.f;
#pragma unroll
  for (int i = 0; i < 4; ++i) {
    v[i] = xr[tid + i * 256];
    s += v[i];
    s2 += v[i] * v[i];
  }
  __shared__ float2 red[256];
  red[tid] = make_float2(s, s2);
  __syncthreads();
  for (int off = 128; off > 0; off >>= 1) {
    if (tid < off) {
      red[tid].x += red[tid + off].x;
      red[tid].y += red[tid + off].y;
    }
    __syncthreads();
  }
  const float mu = red[0].x * (1.0f / 1024.0f);
  const float var = red[0].y * (1.0f / 1024.0f) - mu * mu;
  const float rstd = rsqrtf(var + 1e-5f);
#pragma unroll
  for (int i = 0; i < 4; ++i) {
    int c = tid + i * 256;
    out[(size_t)row * 1024 + c] = (f16)((v[i] - mu) * rstd * w[c] + b[c]);
  }
}

// ---------------------------------------------------------------------------
// Elementwise f32 -> f16 cast (weights)
// ---------------------------------------------------------------------------
__global__ __launch_bounds__(256)
void cast_f32_f16_kernel(const float* __restrict__ in, f16* __restrict__ out,
                         size_t n) {
  size_t i = (size_t)blockIdx.x * blockDim.x + threadIdx.x;
  size_t stride = (size_t)gridDim.x * blockDim.x;
  for (; i < n; i += stride) out[i] = (f16)in[i];
}

// dt slice of dbl (8192 x 96) -> dt_h (8192 x 64) f16
__global__ __launch_bounds__(256)
void cast_dt_kernel(const float* __restrict__ dbl, f16* __restrict__ dt_h) {
  int i = blockIdx.x * 256 + threadIdx.x;  // 8192*64
  int r = i >> 6, c = i & 63;
  dt_h[i] = (f16)dbl[(size_t)r * 96 + c];
}

// ---------------------------------------------------------------------------
// Causal depthwise conv (k=4) over L + bias + SiLU.
// xz_h: (B*L, 4096) f16; channels d in [0,2048). Output xc_h: (B*L, 2048) f16
// ---------------------------------------------------------------------------
__global__ __launch_bounds__(256)
void conv_silu_kernel(const f16* __restrict__ xz_h,
                      const float* __restrict__ conv_w,
                      const float* __restrict__ conv_b,
                      f16* __restrict__ xc_h) {
  int idx = blockIdx.x * 256 + threadIdx.x;  // B*L*2048
  int d = idx & 2047;
  int t = idx >> 11;  // b*L + l
  int l = t & 2047;
  float acc = conv_b[d];
#pragma unroll
  for (int j = 0; j < 4; ++j) {
    int ls = l - 3 + j;
    if (ls >= 0)
      acc += conv_w[d * 4 + j] * (float)xz_h[(size_t)(t - 3 + j) * 4096 + d];
  }
  acc = acc * sigmoidf_fast(acc);  // SiLU
  xc_h[(size_t)t * 2048 + d] = (f16)acc;
}

// ---------------------------------------------------------------------------
// Selective scan. Thread = one (b, d) channel; serial over L; 16-state
// recurrence in registers. B_t/C_t addresses are wave-uniform (blockIdx +
// loop counter), so they scalarize to SMEM loads -- no LDS, no barriers.
// Fused: y = (scan + x*D) * silu(z), stored f16 for the out_proj WMMA.
// ---------------------------------------------------------------------------
__global__ __launch_bounds__(256)
void scan_kernel(const float* __restrict__ delta,  // (B*L, 2048)
                 const f16* __restrict__ xc_h,     // (B*L, 2048)
                 const float* __restrict__ dbl,    // (B*L, 96): dt|B|C
                 const f16* __restrict__ xz_h,     // (B*L, 4096): z at 2048+d
                 const float* __restrict__ A_log,  // (2048, 16)
                 const float* __restrict__ D_param,
                 f16* __restrict__ y_h) {
  const int d = (blockIdx.x & 7) * 256 + threadIdx.x;  // channel
  const int b = blockIdx.x >> 3;                       // batch

  float a[16], h[16];
#pragma unroll
  for (int n = 0; n < 16; ++n) {
    a[n] = -__expf(A_log[d * 16 + n]);
    h[n] = 0.f;
  }
  const float Dp = D_param[d];
  const float* bc_base = dbl + (size_t)b * 2048 * 96 + 64;

  for (int l = 0; l < 2048; ++l) {
    const size_t row = (size_t)b * 2048 + l;
    const float* bc = bc_base + (size_t)l * 96;  // uniform across the wave
    float bcv[32];
#pragma unroll
    for (int n = 0; n < 32; ++n) bcv[n] = bc[n];

    const float dt = delta[row * 2048 + d];
    const float xv = (float)xc_h[row * 2048 + d];
    const float dx = dt * xv;
    float y = 0.f;
#pragma unroll
    for (int n = 0; n < 16; ++n) {
      const float dA = __expf(dt * a[n]);
      h[n] = dA * h[n] + dx * bcv[n];
      y += h[n] * bcv[16 + n];
    }
    y += xv * Dp;
    const float z = (float)xz_h[row * 4096 + 2048 + d];
    y *= z * sigmoidf_fast(z);
    y_h[row * 2048 + d] = (f16)y;
  }
}

// ---------------------------------------------------------------------------
// Host launcher
// ---------------------------------------------------------------------------
extern "C" void kernel_launch(void* const* d_in, const int* in_sizes, int n_in,
                              void* d_out, int out_size, void* d_ws,
                              size_t ws_size, hipStream_t stream) {
  (void)in_sizes; (void)n_in; (void)out_size; (void)ws_size;

  constexpr int BSZ = 4, SEQ = 2048, DM = 1024, DI = 2048;
  constexpr int M = BSZ * SEQ;  // 8192

  const float* x          = (const float*)d_in[0];
  const float* norm_w     = (const float*)d_in[1];
  const float* norm_b     = (const float*)d_in[2];
  const float* in_proj_w  = (const float*)d_in[3];   // (4096, 1024)
  const float* conv_w     = (const float*)d_in[4];   // (2048, 1, 4)
  const float* conv_b     = (const float*)d_in[5];
  const float* x_proj_w   = (const float*)d_in[6];   // (96, 2048)
  const float* dt_proj_w  = (const float*)d_in[7];   // (2048, 64)
  const float* dt_proj_b  = (const float*)d_in[8];
  const float* A_log      = (const float*)d_in[9];   // (2048, 16)
  const float* D_param    = (const float*)d_in[10];
  const float* out_proj_w = (const float*)d_in[11];  // (1024, 2048)
  float* out = (float*)d_out;

  // Workspace carve-out (256B aligned)
  char* base = (char*)d_ws;
  size_t off = 0;
  auto carve = [&](size_t bytes) -> char* {
    char* p = base + off;
    off += (bytes + 255) & ~(size_t)255;
    return p;
  };
  f16*   xn_h  = (f16*)carve((size_t)M * DM * 2);       // 16 MB
  f16*   w1_h  = (f16*)carve((size_t)2 * DI * DM * 2);  // 8 MB
  f16*   xz_h  = (f16*)carve((size_t)M * 2 * DI * 2);   // 64 MB
  f16*   xc_h  = (f16*)carve((size_t)M * DI * 2);       // 32 MB
  f16*   w2_h  = (f16*)carve((size_t)96 * DI * 2);
  float* dbl   = (float*)carve((size_t)M * 96 * 4);     // 3 MB
  f16*   dt_h  = (f16*)carve((size_t)M * 64 * 2);
  f16*   w3_h  = (f16*)carve((size_t)DI * 64 * 2);
  float* delta = (float*)carve((size_t)M * DI * 4);     // 64 MB
  f16*   y_h   = (f16*)carve((size_t)M * DI * 2);       // 32 MB
  f16*   w4_h  = (f16*)carve((size_t)DM * DI * 2);      // 4 MB

  // 1) Weight casts f32 -> f16
  cast_f32_f16_kernel<<<2048, 256, 0, stream>>>(in_proj_w, w1_h,
                                                (size_t)2 * DI * DM);
  cast_f32_f16_kernel<<<256, 256, 0, stream>>>(x_proj_w, w2_h, (size_t)96 * DI);
  cast_f32_f16_kernel<<<256, 256, 0, stream>>>(dt_proj_w, w3_h, (size_t)DI * 64);
  cast_f32_f16_kernel<<<2048, 256, 0, stream>>>(out_proj_w, w4_h,
                                                (size_t)DM * DI);

  // 2) LayerNorm + cast
  layernorm_cast_kernel<<<M, 256, 0, stream>>>(x, norm_w, norm_b, xn_h);

  // 3) in_proj: (8192x1024) x (4096x1024)^T -> xz_h f16
  gemm_wmma_kernel<128, 128, 4, 2, 2, 4, 0>
      <<<dim3(4096 / 128, M / 128), 256, 0, stream>>>(
          xn_h, DM, w1_h, DM, xz_h, M, 2 * DI, DM, nullptr, 0);

  // 4) depthwise causal conv + SiLU -> xc_h f16
  conv_silu_kernel<<<(size_t)M * DI / 256, 256, 0, stream>>>(xz_h, conv_w,
                                                             conv_b, xc_h);

  // 5) x_proj: (8192x2048) x (96x2048)^T -> dbl f32  (BN = 96)
  gemm_wmma_kernel<128, 96, 4, 2, 2, 3, 1>
      <<<dim3(1, M / 128), 256, 0, stream>>>(
          xc_h, DI, w2_h, DI, dbl, M, 96, DI, nullptr, 0);

  // 6) dt slice -> f16
  cast_dt_kernel<<<M * 64 / 256, 256, 0, stream>>>(dbl, dt_h);

  // 7) dt_proj + bias + softplus: (8192x64) x (2048x64)^T -> delta f32
  gemm_wmma_kernel<128, 128, 4, 2, 2, 4, 2>
      <<<dim3(DI / 128, M / 128), 256, 0, stream>>>(
          dt_h, 64, w3_h, 64, delta, M, DI, 64, dt_proj_b, 0);

  // 8) selective scan (+ D skip + SiLU(z) gate) -> y_h f16
  scan_kernel<<<BSZ * (DI / 256), 256, 0, stream>>>(delta, xc_h, dbl, xz_h,
                                                    A_log, D_param, y_h);

  // 9) out_proj + residual: (8192x2048) x (1024x2048)^T + x -> out f32
  gemm_wmma_kernel<128, 128, 4, 2, 2, 4, 3>
      <<<dim3(DM / 128, M / 128), 256, 0, stream>>>(
          y_h, DI, w4_h, DI, out, M, DM, DI, x, DM);
}
